// LocalPoolBaseEncoder_32057635897380
// MI455X (gfx1250) — matrix operand order repaired and checked by reading, not compiled, not run on hardware
//
#include <hip/hip_runtime.h>
#include <hip/hip_bf16.h>

typedef __attribute__((ext_vector_type(16))) _Float16 v16h;
typedef __attribute__((ext_vector_type(8)))  float    v8f;

#define NB   8
#define PP   131072
#define FDIM 32
#define HDIM 128
#define CDIM 32
#define RES  32
#define R3   32768
#define NTILES 65536          /* NB*PP/16 */
#define NROWS  262144         /* NB*R3   */

__device__ __forceinline__ void atomic_max_f32_dev(float* p, float v) {
    unsigned long long a = (unsigned long long)p;
    asm volatile("global_atomic_max_num_f32 %0, %1, off scope:SCOPE_DEV"
                 :: "v"(a), "v"(v) : "memory");
}

__device__ __forceinline__ v8f wmma_f16(v16h a, v16h b, v8f c) {
    return __builtin_amdgcn_wmma_f32_16x16x32_f16(false, a, false, b, (short)0, c,
                                                  false, false);
}

__device__ __forceinline__ void cvt8(v16h& d, int off, float4 a, float4 b) {
    d[off + 0] = (_Float16)a.x; d[off + 1] = (_Float16)a.y;
    d[off + 2] = (_Float16)a.z; d[off + 3] = (_Float16)a.w;
    d[off + 4] = (_Float16)b.x; d[off + 5] = (_Float16)b.y;
    d[off + 6] = (_Float16)b.z; d[off + 7] = (_Float16)b.w;
}

/* ---------------- kernel 1: init pooled grid to -inf ---------------- */
__global__ __launch_bounds__(256) void init_pooled(float4* __restrict__ pooled, int n4) {
    float ninf = __int_as_float(0xFF800000);
    float4 v = {ninf, ninf, ninf, ninf};
    for (int i = blockIdx.x * blockDim.x + threadIdx.x; i < n4;
         i += gridDim.x * blockDim.x)
        pooled[i] = v;
}

/* ------- kernel 2: fused idx + MLP (WMMA) + scatter atomic-max ------- */
__global__ __launch_bounds__(256) void mlp_scatter(
    const float* __restrict__ points, const float* __restrict__ feature,
    const float* __restrict__ fc0_w, const float* __restrict__ fc0_b,
    const float* __restrict__ fc1_w, const float* __restrict__ fc1_b,
    const float* __restrict__ sc_w,
    float* __restrict__ pooled, int* __restrict__ idx_out)
{
    __shared__ _Float16 lds_net[8][16 * 34];                  /* per-wave transpose */
    __shared__ __align__(16) _Float16 lds_w[2][8][32][16];    /* staged B-frags 16KB */
    const int lane = threadIdx.x & 31;
    const int wave = threadIdx.x >> 5;
    const int hi   = lane >> 4;
    const int ln   = lane & 15;
    _Float16* myLds = &lds_net[wave][0];

    /* ---- fc0 B-fragments stay in registers (small) ---- */
    v16h w0[2];
#pragma unroll
    for (int t = 0; t < 2; ++t)
#pragma unroll
        for (int e = 0; e < 16; ++e)
            w0[t][e] = (_Float16)fc0_w[(16 * hi + e) * FDIM + 16 * t + ln];

    /* ---- fc1/sc B-fragments staged to LDS: wave w stages tile t=w ---- */
#pragma unroll
    for (int e = 0; e < 16; ++e) {
        lds_w[0][wave][lane][e] = (_Float16)fc1_w[(16 * hi + e) * HDIM + 16 * wave + ln];
        lds_w[1][wave][lane][e] = (_Float16)sc_w [(16 * hi + e) * HDIM + 16 * wave + ln];
    }
    __syncthreads();

    const float b0c0 = fc0_b[ln];
    const float b0c1 = fc0_b[16 + ln];
    float b1c[8];
#pragma unroll
    for (int t = 0; t < 8; ++t) b1c[t] = fc1_b[16 * t + ln];

    const int wid    = blockIdx.x * 8 + wave;
    const int nwaves = gridDim.x * 8;

    for (int g = wid; g < NTILES; g += nwaves) {
        const int pbase = g * 16;

        /* ---- voxel index (both half-waves compute point ln's index) ---- */
        int idxv;
        {
            const float* pp = points + (size_t)(pbase + ln) * 3;
            float px = pp[0], py = pp[1], pz = pp[2];
            const float inv = 1.0f / 1.021f;   /* 1 + PADDING + 0.001 */
            float qx = fminf(fmaxf((px - 0.5f) * inv + 0.5f, 0.0f), 1.0f - 1e-6f);
            float qy = fminf(fmaxf((py - 0.5f) * inv + 0.5f, 0.0f), 1.0f - 1e-6f);
            float qz = fminf(fmaxf((pz - 0.5f) * inv + 0.5f, 0.0f), 1.0f - 1e-6f);
            int x = (int)(qx * (float)RES);
            int y = (int)(qy * (float)RES);
            int z = (int)(qz * (float)RES);
            idxv = x + RES * (y + RES * z);
        }
        if (lane < 16) idx_out[pbase + ln] = idxv;

        /* ---- feature A-fragments (raw + relu), explicit b128 loads ---- */
        const float* fptr = feature + (size_t)(pbase + ln) * FDIM;
        float4 c0 = *(const float4*)(fptr + 16 * hi);
        float4 c1 = *(const float4*)(fptr + 16 * hi + 4);
        float4 c2 = *(const float4*)(fptr + 16 * hi + 8);
        float4 c3 = *(const float4*)(fptr + 16 * hi + 12);
        v16h a_raw, a_relu;
        cvt8(a_raw, 0, c0, c1);
        cvt8(a_raw, 8, c2, c3);
        c0.x = fmaxf(c0.x, 0.f); c0.y = fmaxf(c0.y, 0.f);
        c0.z = fmaxf(c0.z, 0.f); c0.w = fmaxf(c0.w, 0.f);
        c1.x = fmaxf(c1.x, 0.f); c1.y = fmaxf(c1.y, 0.f);
        c1.z = fmaxf(c1.z, 0.f); c1.w = fmaxf(c1.w, 0.f);
        c2.x = fmaxf(c2.x, 0.f); c2.y = fmaxf(c2.y, 0.f);
        c2.z = fmaxf(c2.z, 0.f); c2.w = fmaxf(c2.w, 0.f);
        c3.x = fmaxf(c3.x, 0.f); c3.y = fmaxf(c3.y, 0.f);
        c3.z = fmaxf(c3.z, 0.f); c3.w = fmaxf(c3.w, 0.f);
        cvt8(a_relu, 0, c0, c1);
        cvt8(a_relu, 8, c2, c3);

        /* ---- fc0: net = relu(relu(x) @ W0 + b0) ---- */
        v8f acc0 = {}, acc1 = {};
        acc0 = wmma_f16(a_relu, w0[0], acc0);
        acc1 = wmma_f16(a_relu, w0[1], acc1);

        /* D-layout (lane=col) -> A-layout (lane=row) via LDS, f16 */
#pragma unroll
        for (int r = 0; r < 8; ++r) {
            int prow = r + 8 * hi;
            myLds[prow * 34 + ln]      = (_Float16)fmaxf(acc0[r] + b0c0, 0.0f);
            myLds[prow * 34 + 16 + ln] = (_Float16)fmaxf(acc1[r] + b0c1, 0.0f);
        }
        asm volatile("s_wait_dscnt 0" ::: "memory");
        v16h a_net;
#pragma unroll
        for (int j = 0; j < 8; ++j) {
            a_net[j]     = myLds[ln * 34 + 16 * hi + j];
            a_net[8 + j] = myLds[ln * 34 + 16 * hi + 8 + j];
        }

        /* ---- per-point voxel rows for the scatter ---- */
        const size_t nR3 = (size_t)(g >> 13) << 15;   /* (point/P)*R3 */
        float* rowp[8];
#pragma unroll
        for (int r = 0; r < 8; ++r) {
            int vox = __shfl(idxv, r + 8 * hi, 32);
            rowp[r] = pooled + (nR3 + (size_t)vox) * HDIM + ln;
        }

        /* ---- fc1 + shortcut (frags from LDS), then scatter atomic max ---- */
#pragma unroll
        for (int t = 0; t < 8; ++t) {
            v8f h;
#pragma unroll
            for (int r = 0; r < 8; ++r) h[r] = b1c[t];
            v16h wf1 = *(const v16h*)&lds_w[0][t][lane][0];
            v16h wfs = *(const v16h*)&lds_w[1][t][lane][0];
            h = wmma_f16(a_net, wf1, h);
            h = wmma_f16(a_raw, wfs, h);
#pragma unroll
            for (int r = 0; r < 8; ++r)
                atomic_max_f32_dev(rowp[r] + 16 * t, h[r]);
        }
    }
}

/* -------- kernel 3: pooled(-inf->0) @ conv_w^T + b, transposed out -------- */
__global__ __launch_bounds__(256) void conv_out(
    const float* __restrict__ pooled, const float* __restrict__ conv_w,
    const float* __restrict__ conv_b, float* __restrict__ out)
{
    __shared__ float lds_out[CDIM * 132];
    const int lane = threadIdx.x & 31;
    const int wave = threadIdx.x >> 5;
    const int hi   = lane >> 4;
    const int ln   = lane & 15;

    /* B-fragments of conv_w^T (K=128 -> 4 chunks, C=32 -> 2 tiles) */
    v16h wb[2][4];
#pragma unroll
    for (int t = 0; t < 2; ++t)
#pragma unroll
        for (int kk = 0; kk < 4; ++kk) {
            const float* wp = conv_w + (size_t)(16 * t + ln) * HDIM + kk * 32 + 16 * hi;
            float4 u0 = *(const float4*)(wp);
            float4 u1 = *(const float4*)(wp + 4);
            float4 u2 = *(const float4*)(wp + 8);
            float4 u3 = *(const float4*)(wp + 12);
            cvt8(wb[t][kk], 0, u0, u1);
            cvt8(wb[t][kk], 8, u2, u3);
        }
    const float bc0 = conv_b[ln];
    const float bc1 = conv_b[16 + ln];

    const size_t rowbase = (size_t)blockIdx.x * 128 + (size_t)wave * 16;
    v8f acc0, acc1;
#pragma unroll
    for (int r = 0; r < 8; ++r) { acc0[r] = bc0; acc1[r] = bc1; }

#pragma unroll
    for (int kk = 0; kk < 4; ++kk) {
        const float* gp = pooled + (rowbase + ln) * HDIM + kk * 32;
        float4 g0 = *(const float4*)(gp + 16 * hi);
        float4 g1 = *(const float4*)(gp + 16 * hi + 4);
        float4 g2 = *(const float4*)(gp + 16 * hi + 8);
        float4 g3 = *(const float4*)(gp + 16 * hi + 12);
        /* empty voxel (-inf) -> 0 */
        g0.x = (g0.x < -1e37f) ? 0.f : g0.x; g0.y = (g0.y < -1e37f) ? 0.f : g0.y;
        g0.z = (g0.z < -1e37f) ? 0.f : g0.z; g0.w = (g0.w < -1e37f) ? 0.f : g0.w;
        g1.x = (g1.x < -1e37f) ? 0.f : g1.x; g1.y = (g1.y < -1e37f) ? 0.f : g1.y;
        g1.z = (g1.z < -1e37f) ? 0.f : g1.z; g1.w = (g1.w < -1e37f) ? 0.f : g1.w;
        g2.x = (g2.x < -1e37f) ? 0.f : g2.x; g2.y = (g2.y < -1e37f) ? 0.f : g2.y;
        g2.z = (g2.z < -1e37f) ? 0.f : g2.z; g2.w = (g2.w < -1e37f) ? 0.f : g2.w;
        g3.x = (g3.x < -1e37f) ? 0.f : g3.x; g3.y = (g3.y < -1e37f) ? 0.f : g3.y;
        g3.z = (g3.z < -1e37f) ? 0.f : g3.z; g3.w = (g3.w < -1e37f) ? 0.f : g3.w;
        v16h a;
        cvt8(a, 0, g0, g1);
        cvt8(a, 8, g2, g3);
        acc0 = wmma_f16(a, wb[0][kk], acc0);
        acc1 = wmma_f16(a, wb[1][kk], acc1);
    }

    /* stage 16x32 tile into LDS as [c][v_local] for contiguous stores */
#pragma unroll
    for (int r = 0; r < 8; ++r) {
        int vl = wave * 16 + r + 8 * hi;
        lds_out[ln        * 132 + vl] = acc0[r];
        lds_out[(16 + ln) * 132 + vl] = acc1[r];
    }
    __syncthreads();

    const int c    = threadIdx.x >> 3;        /* 0..31 */
    const int voff = (threadIdx.x & 7) * 16;  /* 0..112 */
    const size_t row0 = (size_t)blockIdx.x * 128;
    const size_t n    = row0 >> 15;
    const size_t vblk = row0 & (R3 - 1);
    float* op = out + (n * CDIM + (size_t)c) * R3 + vblk + voff;
#pragma unroll
    for (int q = 0; q < 16; q += 4)
        *(float4*)(op + q) = *(float4*)&lds_out[c * 132 + voff + q];
}

extern "C" void kernel_launch(void* const* d_in, const int* in_sizes, int n_in,
                              void* d_out, int out_size, void* d_ws, size_t ws_size,
                              hipStream_t stream) {
    (void)in_sizes; (void)n_in; (void)out_size; (void)ws_size;
    const float* points  = (const float*)d_in[0];
    const float* feature = (const float*)d_in[1];
    const float* fc0_w   = (const float*)d_in[2];
    const float* fc0_b   = (const float*)d_in[3];
    const float* fc1_w   = (const float*)d_in[4];
    const float* fc1_b   = (const float*)d_in[5];
    const float* sc_w    = (const float*)d_in[6];
    const float* conv_w  = (const float*)d_in[7];
    const float* conv_b  = (const float*)d_in[8];

    float* pooled  = (float*)d_ws;                       /* NB*R3*HDIM f32 = 128 MiB */
    int*   idx_out = (int*)d_out;                        /* NB*PP int32 */
    float* grid    = (float*)d_out + (size_t)NB * PP;    /* NB*CDIM*R3 f32 */

    init_pooled<<<2048, 256, 0, stream>>>((float4*)pooled, (NROWS * HDIM) / 4);
    mlp_scatter<<<1024, 256, 0, stream>>>(points, feature, fc0_w, fc0_b,
                                          fc1_w, fc1_b, sc_w, pooled, idx_out);
    conv_out<<<NROWS / 128, 256, 0, stream>>>(pooled, conv_w, conv_b, grid);
}
